// DeepMapping2D_73701638799717
// MI455X (gfx1250) — compile-verified, matrix operand order checked by reading, not run on hardware
//
#include <hip/hip_runtime.h>
#include <hip/hip_bf16.h>
#include <cstdint>
#include <cstddef>

typedef __attribute__((ext_vector_type(8))) int v8i;

#define TOPK        5120
#define GZDIM       1024
#define NBINS       (1 << 20)          // 1024 x 1024 bins per cloud
#define CNT_BLOCKS  64                 // blocks per cloud in count kernel
#define CNT_THREADS 256                // 8 waves (wave32)

// ---------------------------------------------------------------- utilities
__global__ void k_zero_counts(unsigned* __restrict__ kc, int B) {
  int i = blockIdx.x * blockDim.x + threadIdx.x;
  if (i < B) kc[i] = 0u;
}

// zero chunk histogram + init per-cloud mins to +inf
__global__ void k_init(unsigned* __restrict__ hist, size_t nwords,
                       int* __restrict__ minxz, int b0, int chunk) {
  size_t i = (size_t)blockIdx.x * blockDim.x + threadIdx.x;
  size_t stride = (size_t)gridDim.x * blockDim.x;
  for (size_t j = i; j < nwords; j += stride) hist[j] = 0u;
  if (i < (size_t)(2 * chunk)) minxz[2 * b0 + i] = 0x7FFFFFFF;
}

// ------------------------------------------------------- per-cloud min(x,z)
__global__ void k_min(const float* __restrict__ pcd, int* __restrict__ minxz,
                      int b0, int N) {
  int cloud = b0 + blockIdx.y;
  const float2* p = (const float2*)(pcd + (size_t)cloud * N * 2);
  int tid = blockIdx.x * blockDim.x + threadIdx.x;
  int stride = gridDim.x * blockDim.x;
  int mnx = 0x7FFFFFFF, mnz = 0x7FFFFFFF;
  for (int i = tid; i < N; i += stride) {
    float2 v = p[i];
    int ix = (int)rintf(1000.0f * v.x);   // matches jnp.round (RNE)
    int iz = (int)rintf(1000.0f * v.y);
    mnx = min(mnx, ix);
    mnz = min(mnz, iz);
  }
  __shared__ int sx[256], sz[256];
  sx[threadIdx.x] = mnx; sz[threadIdx.x] = mnz;
  __syncthreads();
  for (int s = (int)blockDim.x >> 1; s > 0; s >>= 1) {
    if ((int)threadIdx.x < s) {
      sx[threadIdx.x] = min(sx[threadIdx.x], sx[threadIdx.x + s]);
      sz[threadIdx.x] = min(sz[threadIdx.x], sz[threadIdx.x + s]);
    }
    __syncthreads();
  }
  if (threadIdx.x == 0) {
    atomicMin(&minxz[2 * cloud + 0], sx[0]);
    atomicMin(&minxz[2 * cloud + 1], sz[0]);
  }
}

// ----------------------------------------------------------- histogram scatter
__global__ void k_scatter(const float* __restrict__ pcd, const int* __restrict__ minxz,
                          unsigned* __restrict__ hist, int b0, int N) {
  int c = blockIdx.y;
  int cloud = b0 + c;
  const float2* p = (const float2*)(pcd + (size_t)cloud * N * 2);
  unsigned* h = hist + ((size_t)c << 20);
  int mnx = minxz[2 * cloud + 0];
  int mnz = minxz[2 * cloud + 1];
  int i = blockIdx.x * blockDim.x + threadIdx.x;
  if (i < N) {
    float2 v = p[i];
    int xi = (int)rintf(1000.0f * v.x) - mnx;
    int zi = (int)rintf(1000.0f * v.y) - mnz;
    atomicAdd(&h[xi * GZDIM + zi], 1u);   // no-return atomic (STOREcnt path)
  }
}

// -------------------------------------- occupied-bin count via WMMA IU8 reduce
// Each wave packs 32 lanes x 32 threshold flags (0/1 bytes) into an 8-VGPR IU8
// A operand, multiplies by all-ones B: v_wmma_i32_16x16x64_iu8 accumulates row
// sums in the 16x16 i32 tile. Sum of each lane's 8 D regs == 16x the true ones
// count (16 replicated columns), so block total >> 4 is exact.
__global__ __launch_bounds__(CNT_THREADS)
void k_count(const unsigned* __restrict__ hist, unsigned* __restrict__ kcount,
             int b0, unsigned thr) {
  const int c = blockIdx.y;
  const unsigned* h = hist + ((size_t)c << 20);

  const int binsPerBlock = NBINS / CNT_BLOCKS;                 // 16384
  const int wavesPerBlock = CNT_THREADS / 32;                  // 8
  const int binsPerWave = binsPerBlock / wavesPerBlock;        // 2048
  const int wave = (int)threadIdx.x >> 5;
  const int lane = (int)threadIdx.x & 31;
  const int waveBase = blockIdx.x * binsPerBlock + wave * binsPerWave;

  v8i ones;
  #pragma unroll
  for (int i = 0; i < 8; ++i) ones[i] = 0x01010101;
  v8i acc = (v8i)(0);

  // uniform trip count across the whole block -> EXEC all-ones at every WMMA
  for (int it = 0; it < binsPerWave; it += 32 * 32) {
    const uint4* lp = (const uint4*)(h + waveBase + it + lane * 32);
    v8i a;
    #pragma unroll
    for (int g = 0; g < 8; ++g) {
      uint4 cc = lp[g];
      unsigned f0 = (cc.x >= thr) ? 1u : 0u;
      unsigned f1 = (cc.y >= thr) ? 1u : 0u;
      unsigned f2 = (cc.z >= thr) ? 1u : 0u;
      unsigned f3 = (cc.w >= thr) ? 1u : 0u;
      a[g] = (int)(f0 | (f1 << 8) | (f2 << 16) | (f3 << 24));
    }
    // (sgn_a, A, sgn_b, B, C, reuse_a, reuse_b) -> v_wmma_i32_16x16x64_iu8
    acc = __builtin_amdgcn_wmma_i32_16x16x64_iu8(false, a, false, ones, acc,
                                                 false, false);
  }

  int part = 0;
  #pragma unroll
  for (int i = 0; i < 8; ++i) part += acc[i];

  __shared__ int red[CNT_THREADS];
  red[threadIdx.x] = part;
  __syncthreads();
  for (int s = CNT_THREADS >> 1; s > 0; s >>= 1) {
    if ((int)threadIdx.x < s) red[threadIdx.x] += red[threadIdx.x + s];
    __syncthreads();
  }
  if (threadIdx.x == 0)
    atomicAdd(&kcount[b0 + c], (unsigned)(red[0] >> 4));
}

// ----------------------------------------------- emit step function [B,TOPK,1]
__global__ void k_out(const unsigned* __restrict__ kcount, float* __restrict__ out,
                      int B) {
  int i = blockIdx.x * blockDim.x + threadIdx.x;
  if (i < B * TOPK) {
    int b = i / TOPK, r = i % TOPK;
    unsigned k = kcount[b];
    if (k > (unsigned)TOPK) k = TOPK;
    out[i] = (r < (int)k) ? 1.0f : 0.0f;
  }
}

// ---------------------------------------------------------------- launcher
extern "C" void kernel_launch(void* const* d_in, const int* in_sizes, int n_in,
                              void* d_out, int out_size, void* d_ws, size_t ws_size,
                              hipStream_t stream) {
  const float* pcd = (const float*)d_in[0];
  float* out = (float*)d_out;

  const int B = out_size / TOPK;                       // 64
  const int N = (int)((size_t)in_sizes[0] / (2 * (size_t)B));  // 262144

  // smallest integer t with (float)t / (float)N > 0.0002f  (bit-exact threshold)
  const float fN = (float)N;
  unsigned thr = 0;
  while (!(((float)thr / fN) > 0.0002f)) ++thr;        // -> 53 for N=262144

  // workspace layout: [minxz: B*2 ints][kcount: B u32][hist: chunk * 1M u32]
  int* minxz = (int*)d_ws;
  unsigned* kcount = (unsigned*)((char*)d_ws + (size_t)B * 2 * sizeof(int));
  size_t header = (((size_t)B * 12 + 255) / 256) * 256;
  unsigned* hist = (unsigned*)((char*)d_ws + header);

  size_t histCap = (ws_size > header) ? (ws_size - header) / ((size_t)NBINS * 4) : 0;
  int chunk = (int)histCap;
  if (chunk < 1) chunk = 1;
  if (chunk > B) chunk = B;

  k_zero_counts<<<(B + 255) / 256, 256, 0, stream>>>(kcount, B);

  for (int b0 = 0; b0 < B; b0 += chunk) {
    int cur = (B - b0 < chunk) ? (B - b0) : chunk;
    size_t nwords = (size_t)cur << 20;

    k_init<<<2048, 256, 0, stream>>>(hist, nwords, minxz, b0, cur);

    dim3 gMin(256, cur);
    k_min<<<gMin, 256, 0, stream>>>(pcd, minxz, b0, N);

    dim3 gSc((N + 255) / 256, cur);
    k_scatter<<<gSc, 256, 0, stream>>>(pcd, minxz, hist, b0, N);

    dim3 gCnt(CNT_BLOCKS, cur);
    k_count<<<gCnt, CNT_THREADS, 0, stream>>>(hist, kcount, b0, thr);
  }

  k_out<<<(B * TOPK + 255) / 256, 256, 0, stream>>>(kcount, out, B);
}
